// E_GCL_22703197126945
// MI455X (gfx1250) — matrix operand order repaired and checked by reading, not compiled, not run on hardware
//
#include <hip/hip_runtime.h>
#include <hip/hip_bf16.h>

typedef __attribute__((ext_vector_type(16))) _Float16 v16h;
typedef __attribute__((ext_vector_type(8)))  float    v8f;

#define HID 128
#define EDGED 16

union AccU { v8f v; float f[8]; };
union FragU { v16h h; unsigned int u[8]; };

__device__ __forceinline__ float silu(float x) {
    return x / (1.f + __expf(-x));
}

// ---------------------------------------------------------------------------
// Pack a [K x 128] f32 weight into f16 WMMA B-fragment order, K padded to KP.
// Fragment chunk c = kb*8 + nb holds 32 lanes x 16 halfs:
//   dst[((c*32 + L)*16 + e)] = W[kb*32 + (L>>4)*16 + e][nb*16 + (L&15)]
// so each lane loads one contiguous 32B v16h per WMMA step.
// ---------------------------------------------------------------------------
__global__ void pack_weights_kernel(const float* __restrict__ src,
                                    _Float16* __restrict__ dst,
                                    int Kact, int KP) {
    int t = blockIdx.x * blockDim.x + threadIdx.x;
    int total = KP * HID;
    if (t >= total) return;
    int e    = t & 15;
    int L    = (t >> 4) & 31;
    int frag = t >> 9;
    int nb   = frag & 7;
    int kb   = frag >> 3;
    int k = kb * 32 + (L >> 4) * 16 + e;
    int n = nb * 16 + (L & 15);
    dst[t] = (k < Kact) ? (_Float16)src[k * HID + n] : (_Float16)0.f;
}

__global__ void zero_kernel(float* __restrict__ p, long long n) {
    long long i = (long long)blockIdx.x * blockDim.x + threadIdx.x;
    long long stride = (long long)gridDim.x * blockDim.x;
    for (; i < n; i += stride) p[i] = 0.f;
}

// ---------------------------------------------------------------------------
// One 16x16 output tile of  D = silu?(A[16xK] * B[Kx128] + bias)  per wave.
// A lives in LDS (f16, row stride in halfs), B is pre-packed fragments.
// Wave w owns output columns [16w, 16w+16).
// ---------------------------------------------------------------------------
__device__ __forceinline__ v8f wmma_tile(const _Float16* __restrict__ ldsA,
                                         int rowStride,
                                         const _Float16* __restrict__ Bfrag,
                                         int nkb, int w, int lane, float bias) {
    int g   = lane >> 4;
    int m16 = lane & 15;
    AccU acc;
#pragma unroll
    for (int i = 0; i < 8; ++i) acc.f[i] = bias;

    for (int kb = 0; kb < nkb; ++kb) {
        // A fragment: lane holds row m16, K halves 16*g..16*g+15 in order (contiguous, as packed B)
        FragU a;
        const _Float16* ar = ldsA + m16 * rowStride + kb * 32;
#pragma unroll
        for (int v = 0; v < 8; ++v) {
            int K0 = v * 2 + g * 16;
            a.u[v] = *(const unsigned int*)(const void*)(ar + K0);
        }
        // B fragment: contiguous 32B per lane
        FragU b;
        b.h = *(const v16h*)(const void*)(Bfrag + ((((kb * 8 + w) * 32) + lane) << 4));
        if (kb + 1 < nkb)
            __builtin_prefetch(Bfrag + (((((kb + 1) * 8 + w) * 32) + lane) << 4), 0, 3);
        acc.v = __builtin_amdgcn_wmma_f32_16x16x32_f16(
            false, a.h, false, b.h, (short)0, acc.v, false, false);
    }
    return acc.v;
}

// ---------------------------------------------------------------------------
// Edge kernel: 16 edges per block, 256 threads (8 waves).
//   m  = silu(silu(e_in @ We1 + be1) @ We2 + be2)          -> out_m, agg (+=)
//   wE = silu(m @ Wc1 + bc1) @ Wc2 + bc2                    -> csum, cnt (+=)
// ---------------------------------------------------------------------------
#define KP1 288
#define SP1 296   // LDS row stride (halfs) for e_in tile
#define SPM 136   // LDS row stride (halfs) for 16x128 message tiles

__global__ void __launch_bounds__(256)
edge_kernel(const float* __restrict__ h, const float* __restrict__ coord,
            const float* __restrict__ edge_attr, const int* __restrict__ ei,
            const float* __restrict__ be1, const float* __restrict__ be2,
            const float* __restrict__ bc1, const float* __restrict__ Wc2,
            const float* __restrict__ bc2,
            const _Float16* __restrict__ We1h, const _Float16* __restrict__ We2h,
            const _Float16* __restrict__ Wc1h,
            float* __restrict__ out_m, float* __restrict__ agg,
            float* __restrict__ csum, float* __restrict__ cnt,
            int nE, int nN) {
    __shared__ __align__(16) _Float16 eh[16 * SP1];   // e_in tile, later reused for m tile
    __shared__ __align__(16) _Float16 mh[16 * SPM];   // hidden tile
    __shared__ float s_d[16 * 3];
    __shared__ float s_rad[16];
    __shared__ int   s_row[16];
    __shared__ int   s_col[16];
    __shared__ float s_wsum[16];

    const int tid  = threadIdx.x;
    const int w    = tid >> 5;
    const int lane = tid & 31;
    const int g    = lane >> 4;
    const int m16  = lane & 15;
    const int colw = w * 16 + m16;    // this lane's output column
    const int e0   = blockIdx.x * 16;

    // ---- stage 0: per-edge geometry ----
    if (tid < 16) {
        int eg = e0 + tid;
        int r = 0, c = 0;
        if (eg < nE) { r = ei[eg]; c = ei[nE + eg]; }
        float dx = coord[r * 3 + 0] - coord[c * 3 + 0];
        float dy = coord[r * 3 + 1] - coord[c * 3 + 1];
        float dz = coord[r * 3 + 2] - coord[c * 3 + 2];
        s_d[tid * 3 + 0] = dx; s_d[tid * 3 + 1] = dy; s_d[tid * 3 + 2] = dz;
        s_rad[tid] = dx * dx + dy * dy + dz * dz;
        s_row[tid] = r; s_col[tid] = c;
        s_wsum[tid] = 0.f;
    }
    __syncthreads();

    // ---- stage A: build e_in tile [16 x 288] f16 in LDS ----
    {
        int e = tid >> 4, part = tid & 15;
        int r = s_row[e], c = s_col[e];
#pragma unroll
        for (int j = 0; j < 8; ++j) {
            int cc = part * 8 + j;
            eh[e * SP1 + cc]       = (_Float16)h[r * HID + cc];
            eh[e * SP1 + HID + cc] = (_Float16)h[c * HID + cc];
        }
        for (int cc = 256 + part; cc < KP1; cc += 16) {
            float v = 0.f;
            if (cc == 256)      v = s_rad[e];
            else if (cc < 273)  v = edge_attr[e == 0 && (e0 + e) >= nE ? 0
                                              : ((e0 + e < nE ? e0 + e : 0) * EDGED + (cc - 257))];
            eh[e * SP1 + cc] = (_Float16)v;
        }
    }
    __syncthreads();

    // ---- stage B: m1 = silu(e_in @ We1 + be1) ----
    {
        v8f acc = wmma_tile(eh, SP1, We1h, KP1 / 32, w, lane, be1[colw]);
        AccU a; a.v = acc;
#pragma unroll
        for (int i = 0; i < 8; ++i) {
            int M = i + 8 * g;
            mh[M * SPM + colw] = (_Float16)silu(a.f[i]);
        }
    }
    __syncthreads();

    // ---- stage C: m = silu(m1 @ We2 + be2); emit m, scatter into agg ----
    {
        v8f acc = wmma_tile(mh, SPM, We2h, HID / 32, w, lane, be2[colw]);
        AccU a; a.v = acc;
#pragma unroll
        for (int i = 0; i < 8; ++i) {
            int M = i + 8 * g;
            float mv = silu(a.f[i]);
            eh[M * SPM + colw] = (_Float16)mv;   // reuse eh as f16 m tile
            if (e0 + M < nE) {
                out_m[(long long)(e0 + M) * HID + colw] = mv;
                atomicAdd(&agg[(long long)s_row[M] * HID + colw], mv);
            }
        }
    }
    __syncthreads();

    // ---- stage D: wE = silu(m @ Wc1 + bc1) @ Wc2 + bc2 ----
    {
        v8f acc = wmma_tile(eh, SPM, Wc1h, HID / 32, w, lane, bc1[colw]);
        AccU a; a.v = acc;
        float wc = Wc2[colw];
#pragma unroll
        for (int i = 0; i < 8; ++i) {
            int M = i + 8 * g;
            atomicAdd(&s_wsum[M], silu(a.f[i]) * wc);
        }
    }
    __syncthreads();

    if (tid < 16 && (e0 + tid) < nE) {
        float wv = s_wsum[tid] + bc2[0];
        int node = s_row[tid];
#pragma unroll
        for (int d = 0; d < 3; ++d)
            atomicAdd(&csum[node * 3 + d], s_d[tid * 3 + d] * wv);
        atomicAdd(&cnt[node], 1.f);
    }
}

// ---------------------------------------------------------------------------
// Node kernel: 16 nodes per block.
//   h_out = silu([h ‖ agg] @ Wn1 + bn1) @ Wn2 + bn2
//   coord_out = coord + csum / max(cnt, 1)
// ---------------------------------------------------------------------------
#define SP2 264   // LDS row stride (halfs) for [16 x 256] node-input tile

__global__ void __launch_bounds__(256)
node_kernel(const float* __restrict__ h, const float* __restrict__ coord,
            const float* __restrict__ bn1, const float* __restrict__ bn2,
            const _Float16* __restrict__ Wn1h, const _Float16* __restrict__ Wn2h,
            const float* __restrict__ agg, const float* __restrict__ csum,
            const float* __restrict__ cnt,
            float* __restrict__ out_h, float* __restrict__ out_coord, int nN) {
    __shared__ __align__(16) _Float16 eh2[16 * SP2];
    __shared__ __align__(16) _Float16 mh[16 * SPM];

    const int tid  = threadIdx.x;
    const int w    = tid >> 5;
    const int lane = tid & 31;
    const int g    = lane >> 4;
    const int m16  = lane & 15;
    const int colw = w * 16 + m16;
    const int n0   = blockIdx.x * 16;

    // ---- build [h ‖ agg] tile ----
    {
        int nl = tid >> 4, part = tid & 15;
        int n = (n0 + nl < nN) ? (n0 + nl) : 0;
#pragma unroll
        for (int j = 0; j < 8; ++j) {
            int cc = part * 8 + j;
            eh2[nl * SP2 + cc]       = (_Float16)h[(long long)n * HID + cc];
            eh2[nl * SP2 + HID + cc] = (_Float16)agg[(long long)n * HID + cc];
        }
    }
    __syncthreads();

    // ---- GEMM1: silu(n_in @ Wn1 + bn1) ----
    {
        v8f acc = wmma_tile(eh2, SP2, Wn1h, 256 / 32, w, lane, bn1[colw]);
        AccU a; a.v = acc;
#pragma unroll
        for (int i = 0; i < 8; ++i) {
            int M = i + 8 * g;
            mh[M * SPM + colw] = (_Float16)silu(a.f[i]);
        }
    }
    __syncthreads();

    // ---- GEMM2: @ Wn2 + bn2 (no activation) ----
    {
        v8f acc = wmma_tile(mh, SPM, Wn2h, HID / 32, w, lane, bn2[colw]);
        AccU a; a.v = acc;
#pragma unroll
        for (int i = 0; i < 8; ++i) {
            int M = i + 8 * g;
            if (n0 + M < nN)
                out_h[(long long)(n0 + M) * HID + colw] = a.f[i];
        }
    }

    // ---- coord update ----
    if (tid < 16 && (n0 + tid) < nN) {
        int n = n0 + tid;
        float den = fmaxf(cnt[n], 1.f);
#pragma unroll
        for (int d = 0; d < 3; ++d)
            out_coord[n * 3 + d] = coord[n * 3 + d] + csum[n * 3 + d] / den;
    }
}

// ---------------------------------------------------------------------------
extern "C" void kernel_launch(void* const* d_in, const int* in_sizes, int n_in,
                              void* d_out, int out_size, void* d_ws, size_t ws_size,
                              hipStream_t stream) {
    const float* h         = (const float*)d_in[0];
    const float* coord     = (const float*)d_in[1];
    const float* edge_attr = (const float*)d_in[2];
    const int*   ei        = (const int*)d_in[3];
    const float* We1 = (const float*)d_in[4];  const float* be1 = (const float*)d_in[5];
    const float* We2 = (const float*)d_in[6];  const float* be2 = (const float*)d_in[7];
    const float* Wn1 = (const float*)d_in[8];  const float* bn1 = (const float*)d_in[9];
    const float* Wn2 = (const float*)d_in[10]; const float* bn2 = (const float*)d_in[11];
    const float* Wc1 = (const float*)d_in[12]; const float* bc1 = (const float*)d_in[13];
    const float* Wc2 = (const float*)d_in[14]; const float* bc2 = (const float*)d_in[15];

    const int nN = in_sizes[0] / HID;
    const int nE = in_sizes[3] / 2;

    // workspace layout: f32 accumulators, then 32B-aligned f16 weight fragments
    float* agg  = (float*)d_ws;                  // nN * 128
    float* csum = agg + (long long)nN * HID;     // nN * 3
    float* cnt  = csum + (long long)nN * 3;      // nN
    long long zeroN = (long long)nN * (HID + 3 + 1);
    // align f16 region to 32 bytes
    char* wsc = (char*)(cnt + nN);
    size_t off = (size_t)wsc & 31; if (off) wsc += 32 - off;
    _Float16* We1h = (_Float16*)wsc;             // 288*128
    _Float16* We2h = We1h + KP1 * HID;           // 128*128
    _Float16* Wc1h = We2h + HID * HID;           // 128*128
    _Float16* Wn1h = Wc1h + HID * HID;           // 256*128
    _Float16* Wn2h = Wn1h + 256 * HID;           // 128*128

    float* out_h     = (float*)d_out;
    float* out_coord = out_h + (long long)nN * HID;
    float* out_m     = out_coord + (long long)nN * 3;

    // zero accumulators (every call: graph replays do not re-zero for us)
    zero_kernel<<<2048, 256, 0, stream>>>(agg, zeroN);

    // pack weights to f16 fragments
    pack_weights_kernel<<<(KP1 * HID + 255) / 256, 256, 0, stream>>>(We1, We1h, 273, KP1);
    pack_weights_kernel<<<(HID * HID + 255) / 256, 256, 0, stream>>>(We2, We2h, HID, HID);
    pack_weights_kernel<<<(HID * HID + 255) / 256, 256, 0, stream>>>(Wc1, Wc1h, HID, HID);
    pack_weights_kernel<<<(256 * HID + 255) / 256, 256, 0, stream>>>(Wn1, Wn1h, 256, 256);
    pack_weights_kernel<<<(HID * HID + 255) / 256, 256, 0, stream>>>(Wn2, Wn2h, HID, HID);

    edge_kernel<<<(nE + 15) / 16, 256, 0, stream>>>(
        h, coord, edge_attr, ei, be1, be2, bc1, Wc2, bc2,
        We1h, We2h, Wc1h, out_m, agg, csum, cnt, nE, nN);

    node_kernel<<<(nN + 15) / 16, 256, 0, stream>>>(
        h, coord, bn1, bn2, Wn1h, Wn2h, agg, csum, cnt,
        out_h, out_coord, nN);
}
